// SelfAttention_67585605370598
// MI455X (gfx1250) — compile-verified, matrix operand order checked by reading, not run on hardware
//
#include <hip/hip_runtime.h>
#include <hip/hip_bf16.h>

// ---------------------------------------------------------------------------
// Self-attention forward for MI455X (gfx1250), bf16 WMMA path with
// double-buffered async global->LDS staging in the projection GEMMs.
// d_out = [ out (8*1024*512 f32) | attn (8*8*1024*1024 f32) ]
// ---------------------------------------------------------------------------

typedef __bf16 bf16;
typedef __attribute__((ext_vector_type(16))) __bf16 v16bf;
typedef __attribute__((ext_vector_type(8)))  __bf16 v8bf;
typedef __attribute__((ext_vector_type(8)))  float  v8f;

#define B_      8
#define S_      1024
#define DIM_    512
#define HEADS_  8
#define DHEAD_  64
#define INNER_  512          // HEADS_*DHEAD_
#define N_QKV   1536         // 3*INNER_
#define ROWS_   (B_*S_)      // 8192
#define SCALE_  0.125f       // 64^-0.5
#define LDSROW  40           // padded LDS row stride in bf16 (80B: conflict-free)

__device__ __forceinline__ v8f wmma_bf16(v16bf a, v16bf b, v8f c) {
  return __builtin_amdgcn_wmma_f32_16x16x32_bf16(
      /*neg_a=*/false, a, /*neg_b=*/false, b,
      /*c_mod=*/(short)0, c, /*reuse_a=*/false, /*reuse_b=*/false);
}

// WMMA A/B operand (16 bf16) from a row pointer at the K-step base.
// Per-lane chunks: khalf*8..+7 and 16+khalf*8..+7 (two contiguous 16B loads).
__device__ __forceinline__ v16bf load_op(const bf16* __restrict__ row, int khalf) {
  v8bf lo = *(const v8bf*)(row + khalf * 8);
  v8bf hi = *(const v8bf*)(row + 16 + khalf * 8);
  v16bf r;
#pragma unroll
  for (int i = 0; i < 8; ++i) { r[i] = lo[i]; r[i + 8] = hi[i]; }
  return r;
}

// Raw LDS byte offset of a pointer into a __shared__ array (generic -> AS3).
__device__ __forceinline__ unsigned lds_addr(const void* p) {
  return (unsigned)(unsigned long long)
      (__attribute__((address_space(3))) const void*)p;
}

// Async DMA of 16B global -> LDS (tracked by ASYNCcnt).
__device__ __forceinline__ void async_ld_b128(unsigned lds_byte_off, const void* gsrc) {
  asm volatile("global_load_async_to_lds_b128 %0, %1, off"
               :: "v"(lds_byte_off), "v"((unsigned long long)gsrc)
               : "memory");
}

#define WAIT_ASYNC_5() asm volatile("s_wait_asynccnt 0x5" ::: "memory")
#define WAIT_ASYNC_0() asm volatile("s_wait_asynccnt 0x0" ::: "memory")

// ---------------------------------------------------------------------------
// Elementwise cast f32 -> bf16 (for x)
// ---------------------------------------------------------------------------
__global__ void cast_f32_bf16(const float* __restrict__ in, bf16* __restrict__ out, int n) {
  int i = blockIdx.x * blockDim.x + threadIdx.x;
  if (i < n) out[i] = (bf16)in[i];
}

// Transpose + cast: in[K,N] f32 -> out[N,K] bf16.
__global__ void transpose_cast(const float* __restrict__ in, bf16* __restrict__ out,
                               int K, int N) {
  int i = blockIdx.x * blockDim.x + threadIdx.x;
  if (i < K * N) {
    int k = i / N, n = i % N;
    out[n * K + k] = (bf16)in[i];
  }
}

// ---------------------------------------------------------------------------
// GEMM core: 256 threads = 8 waves as 2(M) x 4(N); wave tile 32x64
// (acc[2][4]); WG tile 64x256; K = 512 in steps of 32.
// A[.,512] and Bt[.,512] staged to LDS by double-buffered async b128 copies
// (5 per thread per stage). K-loop unrolled by 2 stages with literal stage
// indices; last two K-steps peeled (no conditional issue in the hot loop).
// ---------------------------------------------------------------------------
__device__ __forceinline__ void gemm_issue(const bf16* aSrc, const bf16* bSrc,
                                           unsigned aDst, unsigned bDst, int k) {
  async_ld_b128(aDst, aSrc + k);
#pragma unroll
  for (int c = 0; c < 4; ++c) async_ld_b128(bDst + c * 16, bSrc + k + c * 8);
}

__device__ __forceinline__ void gemm_compute(const bf16* Als, const bf16* Bls, int s,
                                             int wm, int wn, int l15, int khalf,
                                             v8f acc[2][4]) {
  const bf16* a0 = Als + (s * 64 + wm * 32 + l15) * LDSROW;
  const bf16* bb = Bls + (s * 256 + wn * 64 + l15) * LDSROW;
  v16bf av0 = load_op(a0, khalf);
  v16bf av1 = load_op(a0 + 16 * LDSROW, khalf);
  v16bf bv0 = load_op(bb + 0 * 16 * LDSROW, khalf);
  v16bf bv1 = load_op(bb + 1 * 16 * LDSROW, khalf);
  v16bf bv2 = load_op(bb + 2 * 16 * LDSROW, khalf);
  v16bf bv3 = load_op(bb + 3 * 16 * LDSROW, khalf);
  acc[0][0] = wmma_bf16(av0, bv0, acc[0][0]);
  acc[0][1] = wmma_bf16(av0, bv1, acc[0][1]);
  acc[0][2] = wmma_bf16(av0, bv2, acc[0][2]);
  acc[0][3] = wmma_bf16(av0, bv3, acc[0][3]);
  acc[1][0] = wmma_bf16(av1, bv0, acc[1][0]);
  acc[1][1] = wmma_bf16(av1, bv1, acc[1][1]);
  acc[1][2] = wmma_bf16(av1, bv2, acc[1][2]);
  acc[1][3] = wmma_bf16(av1, bv3, acc[1][3]);
}

__device__ __forceinline__ void gemm_core(const bf16* __restrict__ A,
                                          const bf16* __restrict__ Bt,
                                          int mblk, int nblk,
                                          bf16* Als,   // [2][64 * LDSROW]
                                          bf16* Bls,   // [2][256 * LDSROW]
                                          v8f acc[2][4]) {
  const int tid   = threadIdx.x;
  const int lane  = tid & 31;
  const int wave  = tid >> 5;
  const int wm    = wave >> 2;          // 0..1
  const int wn    = wave & 3;           // 0..3
  const int khalf = lane >> 4;
  const int l15   = lane & 15;

  // Per-thread copy assignments.
  const int arow = tid >> 2, ac = tid & 3;                 // A: 64 rows x 4 chunks
  const bf16* aSrc = A + (size_t)(mblk + arow) * 512 + ac * 8;
  const bf16* bSrc = Bt + (size_t)(nblk + tid) * 512;      // B: 256 rows
  const unsigned aDst0 = lds_addr(Als + arow * LDSROW + ac * 8);
  const unsigned aDst1 = lds_addr(Als + (64 + arow) * LDSROW + ac * 8);
  const unsigned bDst0 = lds_addr(Bls + tid * LDSROW);
  const unsigned bDst1 = lds_addr(Bls + (256 + tid) * LDSROW);

  // Prologue: stage 0 <- k=0.
  gemm_issue(aSrc, bSrc, aDst0, bDst0, 0);

  int k0 = 0;
#pragma unroll 1
  for (int it = 0; it < 7; ++it) {       // K-steps 0..13 (k = 0..416)
    gemm_issue(aSrc, bSrc, aDst1, bDst1, k0 + 32);
    WAIT_ASYNC_5();
    __syncthreads();
    gemm_compute(Als, Bls, 0, wm, wn, l15, khalf, acc);
    __syncthreads();
    gemm_issue(aSrc, bSrc, aDst0, bDst0, k0 + 64);
    WAIT_ASYNC_5();
    __syncthreads();
    gemm_compute(Als, Bls, 1, wm, wn, l15, khalf, acc);
    __syncthreads();
    k0 += 64;
  }
  // Peeled tail: k = 448 (stage 0, prefetch 480 into stage 1), then k = 480.
  gemm_issue(aSrc, bSrc, aDst1, bDst1, 480);
  WAIT_ASYNC_5();
  __syncthreads();
  gemm_compute(Als, Bls, 0, wm, wn, l15, khalf, acc);
  WAIT_ASYNC_0();
  __syncthreads();
  gemm_compute(Als, Bls, 1, wm, wn, l15, khalf, acc);
}

// ---------------------------------------------------------------------------
// QKV GEMM: C[8192,1536] = Xbf @ WqkvT^T; epilogue scatters into per-head
// Q (scaled), K, and V-transposed bf16 buffers. The Q/K/V section is uniform
// per workgroup (256-column span never crosses a 512 boundary), so the
// 3-way branch is a single scalar decision.
// ---------------------------------------------------------------------------
__global__ __launch_bounds__(256) void qkv_gemm(
    const bf16* __restrict__ A,   // [8192,512]
    const bf16* __restrict__ Bt,  // [1536,512]
    bf16* __restrict__ Qb,        // [B,H,S,64]
    bf16* __restrict__ Kb,        // [B,H,S,64]
    bf16* __restrict__ Vt) {      // [B,H,64,S]
  __shared__ bf16 Als[2 * 64 * LDSROW];
  __shared__ bf16 Bls[2 * 256 * LDSROW];
  const int mblk = blockIdx.y * 64;
  const int nblk = blockIdx.x * 256;

  v8f acc[2][4] = {};
  gemm_core(A, Bt, mblk, nblk, Als, Bls, acc);

  const int lane = threadIdx.x & 31;
  const int wave = threadIdx.x >> 5;
  const int wm = wave >> 2, wn = wave & 3;
  const int l15 = lane & 15;

  const int sec   = nblk >> 9;                  // 0=Q 1=K 2=V (uniform per WG)
  const int nbase = (nblk & 511) + wn * 64;     // column within section

  if (sec == 0) {
#pragma unroll
    for (int ms = 0; ms < 2; ++ms) {
      const int rbase = mblk + wm * 32 + ms * 16;
#pragma unroll
      for (int i = 0; i < 8; ++i) {
        int r = rbase + ((lane < 16) ? i : 8 + i);
        int bb = r >> 10, ss = r & 1023;
#pragma unroll
        for (int t = 0; t < 4; ++t) {
          int n = nbase + t * 16 + l15, h = n >> 6, d = n & 63;
          Qb[(((size_t)bb * HEADS_ + h) * S_ + ss) * DHEAD_ + d] =
              (bf16)(acc[ms][t][i] * SCALE_);
        }
      }
    }
  } else if (sec == 1) {
#pragma unroll
    for (int ms = 0; ms < 2; ++ms) {
      const int rbase = mblk + wm * 32 + ms * 16;
#pragma unroll
      for (int i = 0; i < 8; ++i) {
        int r = rbase + ((lane < 16) ? i : 8 + i);
        int bb = r >> 10, ss = r & 1023;
#pragma unroll
        for (int t = 0; t < 4; ++t) {
          int n = nbase + t * 16 + l15, h = n >> 6, d = n & 63;
          Kb[(((size_t)bb * HEADS_ + h) * S_ + ss) * DHEAD_ + d] = (bf16)acc[ms][t][i];
        }
      }
    }
  } else {
#pragma unroll
    for (int ms = 0; ms < 2; ++ms) {
      const int rbase = mblk + wm * 32 + ms * 16;
#pragma unroll
      for (int i = 0; i < 8; ++i) {
        int r = rbase + ((lane < 16) ? i : 8 + i);
        int bb = r >> 10, ss = r & 1023;
#pragma unroll
        for (int t = 0; t < 4; ++t) {
          int n = nbase + t * 16 + l15, h = n >> 6, d = n & 63;
          Vt[(((size_t)bb * HEADS_ + h) * DHEAD_ + d) * S_ + ss] = (bf16)acc[ms][t][i];
        }
      }
    }
  }
}

// ---------------------------------------------------------------------------
// Output projection: out[8192,512] = Obf @ WoutT^T + bias (f32 to d_out).
// ---------------------------------------------------------------------------
__global__ __launch_bounds__(256) void proj_gemm(
    const bf16* __restrict__ A,   // [8192,512]
    const bf16* __restrict__ Bt,  // [512,512]
    const float* __restrict__ bias,
    float* __restrict__ out) {
  __shared__ bf16 Als[2 * 64 * LDSROW];
  __shared__ bf16 Bls[2 * 256 * LDSROW];
  const int mblk = blockIdx.y * 64;
  const int nblk = blockIdx.x * 256;

  v8f acc[2][4] = {};
  gemm_core(A, Bt, mblk, nblk, Als, Bls, acc);

  const int lane = threadIdx.x & 31;
  const int wave = threadIdx.x >> 5;
  const int wm = wave >> 2, wn = wave & 3;
  const int l15 = lane & 15;

#pragma unroll
  for (int ms = 0; ms < 2; ++ms) {
    const int rbase = mblk + wm * 32 + ms * 16;
#pragma unroll
    for (int i = 0; i < 8; ++i) {
      int r = rbase + ((lane < 16) ? i : 8 + i);
#pragma unroll
      for (int t = 0; t < 4; ++t) {
        int n = nblk + wn * 64 + t * 16 + l15;
        out[(size_t)r * DIM_ + n] = acc[ms][t][i] + bias[n];
      }
    }
  }
}

// ---------------------------------------------------------------------------
// Attention: one 128-thread block (4 waves) per (b, h, 16-query tile).
// Phase 1: S = QK^T (scale folded into Q) -> 64KB LDS strip [16][1024] f32.
// Phase 2: row softmax (shfl reductions, 8 lanes/row), probs -> global attn
//          and back into LDS in place.
// Phase 3: O = P @ V; P read from LDS (cvt f32->bf16 in-register), V^T rows
//          contiguous; wave w covers d-columns 16w..16w+15. O stored bf16.
// ---------------------------------------------------------------------------
__global__ __launch_bounds__(128) void attn_kernel(
    const bf16* __restrict__ Qb, const bf16* __restrict__ Kb,
    const bf16* __restrict__ Vt, float* __restrict__ attn,
    bf16* __restrict__ O) {
  __shared__ float sc[16 * S_];   // 64 KB

  const int gid  = blockIdx.x;
  const int qb   = gid & 63;
  const int h    = (gid >> 6) & 7;
  const int b    = gid >> 9;
  const int tid  = threadIdx.x;
  const int lane = tid & 31;
  const int wave = tid >> 5;
  const int l15  = lane & 15;
  const int khalf = lane >> 4;

  const size_t hb = (size_t)(b * HEADS_ + h);
  const bf16* Qh = Qb + (hb * S_ + qb * 16) * DHEAD_;
  const bf16* Kh = Kb + hb * S_ * DHEAD_;
  const bf16* Vh = Vt + hb * DHEAD_ * S_;

  // ---- Phase 1: scores ----
  const bf16* qrow = Qh + l15 * DHEAD_;
  v16bf qa0 = load_op(qrow, khalf);        // d = 0..31
  v16bf qa1 = load_op(qrow + 32, khalf);   // d = 32..63
  for (int t = 0; t < 16; ++t) {
    int kb = wave + 4 * t;                 // key block 0..63
    const bf16* krow = Kh + (size_t)(kb * 16 + l15) * DHEAD_;
    __builtin_prefetch((const void*)(krow + 64 * DHEAD_), 0, 1);
    v16bf k0 = load_op(krow, khalf);
    v16bf k1 = load_op(krow + 32, khalf);
    v8f acc = {};
    acc = wmma_bf16(qa0, k0, acc);
    acc = wmma_bf16(qa1, k1, acc);
#pragma unroll
    for (int i = 0; i < 8; ++i) {
      int r = (lane < 16) ? i : 8 + i;
      sc[r * S_ + kb * 16 + l15] = acc[i];
    }
  }
  __syncthreads();

  // ---- Phase 2: softmax ----
  const int row = tid >> 3;     // 0..15
  const int seg = tid & 7;      // 8 lanes per row (contiguous in a wave)
  float* srow = sc + row * S_ + seg * 128;
  float m = -3.0e38f;
#pragma unroll 4
  for (int i = 0; i < 128; ++i) m = fmaxf(m, srow[i]);
  m = fmaxf(m, __shfl_xor(m, 1));
  m = fmaxf(m, __shfl_xor(m, 2));
  m = fmaxf(m, __shfl_xor(m, 4));
  float ssum = 0.0f;
#pragma unroll 4
  for (int i = 0; i < 128; ++i) { float e = __expf(srow[i] - m); srow[i] = e; ssum += e; }
  ssum += __shfl_xor(ssum, 1);
  ssum += __shfl_xor(ssum, 2);
  ssum += __shfl_xor(ssum, 4);
  const float inv = 1.0f / ssum;
  float* arow = attn + ((hb * S_ + qb * 16 + row) * (size_t)S_) + seg * 128;
#pragma unroll 4
  for (int i = 0; i < 128; ++i) { float p = srow[i] * inv; srow[i] = p; arow[i] = p; }
  __syncthreads();

  // ---- Phase 3: O = P @ V ----
  v8f oacc = {};
  const bf16*  vrow = Vh + (size_t)(wave * 16 + l15) * S_;   // row = d column
  const float* prow = sc + l15 * S_;                         // row = query
  for (int kk = 0; kk < S_; kk += 32) {
    v8f p0 = *(const v8f*)(prow + kk + khalf * 8);
    v8f p1 = *(const v8f*)(prow + kk + 16 + khalf * 8);
    v16bf pa;
#pragma unroll
    for (int i = 0; i < 8; ++i) { pa[i] = (bf16)p0[i]; pa[i + 8] = (bf16)p1[i]; }
    v16bf vb = load_op(vrow + kk, khalf);
    oacc = wmma_bf16(pa, vb, oacc);
  }
#pragma unroll
  for (int i = 0; i < 8; ++i) {
    int r = b * S_ + qb * 16 + ((lane < 16) ? i : 8 + i);
    O[(size_t)r * INNER_ + h * DHEAD_ + wave * 16 + l15] = (bf16)oacc[i];
  }
}

// ---------------------------------------------------------------------------
// Launch
// ---------------------------------------------------------------------------
extern "C" void kernel_launch(void* const* d_in, const int* in_sizes, int n_in,
                              void* d_out, int out_size, void* d_ws, size_t ws_size,
                              hipStream_t stream) {
  (void)in_sizes; (void)n_in; (void)out_size; (void)ws_size;

  const float* x    = (const float*)d_in[0];   // [8,1024,512]
  const float* Wqkv = (const float*)d_in[1];   // [512,1536]
  const float* Wout = (const float*)d_in[2];   // [512,512]
  const float* bout = (const float*)d_in[3];   // [512]

  float* out_p  = (float*)d_out;                         // 8*1024*512
  float* attn_p = out_p + (size_t)ROWS_ * DIM_;          // 8*8*1024*1024

  // Workspace carve-up.
  char* ws = (char*)d_ws;
  bf16* Xbf   = (bf16*)ws;  ws += (size_t)ROWS_ * DIM_ * 2;                 // 8 MB
  bf16* WqkvT = (bf16*)ws;  ws += (size_t)N_QKV * DIM_ * 2;                 // 1.5 MB
  bf16* WoutT = (bf16*)ws;  ws += (size_t)DIM_ * INNER_ * 2;                // 0.5 MB
  bf16* Qb    = (bf16*)ws;  ws += (size_t)B_ * HEADS_ * S_ * DHEAD_ * 2;    // 8 MB
  bf16* Kb    = (bf16*)ws;  ws += (size_t)B_ * HEADS_ * S_ * DHEAD_ * 2;    // 8 MB
  bf16* Vt    = (bf16*)ws;  ws += (size_t)B_ * HEADS_ * S_ * DHEAD_ * 2;    // 8 MB
  bf16* Obf   = (bf16*)ws;  ws += (size_t)ROWS_ * INNER_ * 2;               // 8 MB

  {
    int n = ROWS_ * DIM_;
    cast_f32_bf16<<<(n + 255) / 256, 256, 0, stream>>>(x, Xbf, n);
  }
  {
    int n = DIM_ * N_QKV;
    transpose_cast<<<(n + 255) / 256, 256, 0, stream>>>(Wqkv, WqkvT, DIM_, N_QKV);
  }
  {
    int n = INNER_ * DIM_;
    transpose_cast<<<(n + 255) / 256, 256, 0, stream>>>(Wout, WoutT, INNER_, DIM_);
  }
  {
    dim3 grid(N_QKV / 256, ROWS_ / 64);   // (6, 128)
    qkv_gemm<<<grid, 256, 0, stream>>>(Xbf, WqkvT, Qb, Kb, Vt);
  }
  {
    int nblocks = B_ * HEADS_ * (S_ / 16);   // 4096
    attn_kernel<<<nblocks, 128, 0, stream>>>(Qb, Kb, Vt, attn_p, Obf);
  }
  {
    dim3 grid(DIM_ / 256, ROWS_ / 64);    // (2, 128)
    proj_gemm<<<grid, 256, 0, stream>>>(Obf, WoutT, bout, out_p);
  }
}
